// SSCNNPredictor_30666066493783
// MI455X (gfx1250) — compile-verified, matrix operand order checked by reading, not run on hardware
//
#include <hip/hip_runtime.h>

#define B_   2
#define T_   128
#define L_   256
#define C_   768
#define BBN_ 16
#define KC_  (C_ / 32)   // 24 k-chunks
#define NT_  (C_ / 16)   // 48 n-tiles
#define RW_  4           // rows per gram_stem block

typedef __attribute__((ext_vector_type(16))) __bf16        v16bf;
typedef __attribute__((ext_vector_type(8)))  float         v8f;
typedef __attribute__((ext_vector_type(4)))  unsigned int  v4u;

union FragBF {
  v16bf v;
  unsigned short u[16];
  v4u q[2];          // two 16-byte halves (K groups {base..+7},{base+16..+7})
};

__device__ __forceinline__ unsigned short f2bf(float f) {
  unsigned int x = __float_as_uint(f);
  x = (x + 0x7FFFu + ((x >> 16) & 1u)) >> 16;   // round-to-nearest-even
  return (unsigned short)x;
}
__device__ __forceinline__ float bf2f(unsigned short h) {
  return __uint_as_float(((unsigned int)h) << 16);
}
// CDNA5 16-bit A/B fragment K index: element e (0..15), lane half hl (0/1)
__device__ __forceinline__ int frag_k(int e, int hl) {
  return (e < 8 ? e : e + 8) + 8 * hl;
}

// ---------------- kernel 0: token->nucleotide remap, fp32 -> bf16(pairs) ---
__global__ void prep_map(const float* __restrict__ hidden,
                         const int* __restrict__ ptl,
                         unsigned int* __restrict__ mapb16_u32) {
  int b = blockIdx.x;
  __shared__ int toksh[L_];
  int tid = threadIdx.x;
  if (tid < L_) toksh[tid] = T_ - 1;
  __syncthreads();
  if (tid == 0) {  // searchsorted(cumsum(ptl), pos, 'right'), clipped
    int pos = 0;
    for (int t = 0; t < T_ && pos < L_; ++t) {
      int nxt = pos + ptl[b * T_ + t];
      int hi = nxt < L_ ? nxt : L_;
      for (int l = pos; l < hi; ++l) toksh[l] = t;
      pos = nxt;
    }
  }
  __syncthreads();
  const int NP = L_ * C_ / 2;   // pairs
  for (int idx = tid; idx < NP; idx += blockDim.x) {
    int el = idx * 2;
    int l = el / C_, c = el - l * C_;
    const float* src = &hidden[(b * T_ + toksh[l]) * C_ + c];
    unsigned int lo16 = f2bf(src[0]);
    unsigned int hi16 = f2bf(src[1]);
    mapb16_u32[b * NP + idx] = lo16 | (hi16 << 16);
  }
}

// ---------------- kernel 0b: W_nuc fp32 -> bf16, swizzled to fragment order
__global__ void swz_wnuc(const float* __restrict__ w,
                         unsigned short* __restrict__ o) {
  int idx = blockIdx.x * blockDim.x + threadIdx.x;     // one ushort each
  int e    = idx & 15;
  int lane = (idx >> 4) & 31;
  int t    = idx >> 9;                                 // (n*KC_+kc)*NT_+ntile
  int ntile = t % NT_;
  int kc    = (t / NT_) % KC_;
  int n     = t / (NT_ * KC_);
  int hl = lane >> 4, lo = lane & 15;
  int k   = kc * 32 + frag_k(e, hl);
  int col = ntile * 16 + lo;
  o[idx] = f2bf(w[(n * C_ + k) * C_ + col]);
}

// ---------------- kernel 1: 5-way nucleotide GEMM + select -----------------
__global__ void __launch_bounds__(128) nuc_gemm(
    const unsigned short* __restrict__ mapb16,
    const unsigned short* __restrict__ wnb_sw,
    const float* __restrict__ b_nuc,
    const float* __restrict__ wmask,
    const int* __restrict__ nuc_ids,
    unsigned short* __restrict__ h,
    unsigned short* __restrict__ hT) {
  int b = blockIdx.z;
  int mtile = blockIdx.y;
  int wave = threadIdx.x >> 5;
  int lane = threadIdx.x & 31;
  int ntile = blockIdx.x * 4 + wave;
  int hl = lane >> 4, lo = lane & 15;
  int dcol = ntile * 16 + lo;
  const unsigned short* arow = &mapb16[(b * L_ + mtile * 16 + lo) * C_];
  v8f acc[5] = {};
  for (int kc = 0; kc < KC_; ++kc) {
    FragBF a;
    int k0 = kc * 32 + 8 * hl;
    a.q[0] = *(const v4u*)&arow[k0];
    a.q[1] = *(const v4u*)&arow[k0 + 16];
#pragma unroll
    for (int n = 0; n < 5; ++n) {
      const unsigned short* base =
          &wnb_sw[(((size_t)n * KC_ + kc) * NT_ + ntile) * 512 + lane * 16];
      if (n == 0 && kc + 1 < KC_)
        __builtin_prefetch(&wnb_sw[(((size_t)n * KC_ + kc + 1) * NT_ + ntile) * 512], 0, 1);
      FragBF bb;
      bb.q[0] = *(const v4u*)&base[0];
      bb.q[1] = *(const v4u*)&base[8];
      acc[n] = __builtin_amdgcn_wmma_f32_16x16x32_bf16(
          false, a.v, false, bb.v, (short)0, acc[n], false, false);
    }
  }
#pragma unroll
  for (int v = 0; v < 8; ++v) {
    int m = mtile * 16 + v + 8 * hl;
    int n = nuc_ids[b * L_ + m];
    float val;
    switch (n) {
      case 0: val = acc[0][v]; break;
      case 1: val = acc[1][v]; break;
      case 2: val = acc[2][v]; break;
      case 3: val = acc[3][v]; break;
      default: val = acc[4][v]; break;
    }
    val += b_nuc[n * C_ + dcol];
    val *= wmask[b * L_ + m];
    unsigned short hv = f2bf(val);
    h[(b * L_ + m) * C_ + dcol] = hv;
    hT[(b * C_ + dcol) * L_ + m] = hv;
  }
}

// ---------------- kernel 2: fused Gram x stem conv, RW_=4 rows per block ---
// y0[o,i,j] = relu( sum_{s,k} w_in[o,k,s]*h[i+di-1,k]*h[j+dj-1,k] + b_in[o] )
// B panel (hT) is shared by all rows; A = (w slice) * (h row) formed in regs
// with packed bf16 multiplies that co-execute with the WMMA pipe.
__global__ void __launch_bounds__(512) gram_stem(
    const unsigned short* __restrict__ h,
    const unsigned short* __restrict__ hT,
    const float* __restrict__ w_in,
    const float* __restrict__ b_in,
    float* __restrict__ y0,
    unsigned short* __restrict__ x16) {
  int b = blockIdx.y;
  int i0 = blockIdx.x * RW_;
  int tid = threadIdx.x;
  int wave = tid >> 5, lane = tid & 31;
  int hl = lane >> 4, lo = lane & 15;
  int jcol = wave * 16 + lo;
  // transposed hT panel: row = j+1 (0..257), col = kk; stride 40 keeps 16B align
  __shared__ unsigned short hTpT[258][40];
  __shared__ unsigned short wP[9][16][32];     // w_in slice for this k-chunk
  __shared__ unsigned short hr[RW_ + 2][32];   // h rows i0-1 .. i0+RW_
  v8f acc[RW_] = {};
  for (int kc = 0; kc < KC_; ++kc) {
    __syncthreads();
    for (int idx = tid; idx < 32 * 258; idx += 512) {
      int kk = idx / 258, jj = idx - kk * 258;
      int j = jj - 1;
      hTpT[jj][kk] = (j >= 0 && j < L_)
          ? hT[(b * C_ + kc * 32 + kk) * L_ + j] : (unsigned short)0;
      if (kc + 1 < KC_ && idx < 32)
        __builtin_prefetch(&hT[(b * C_ + (kc + 1) * 32 + idx) * L_], 0, 1);
    }
    for (int idx = tid; idx < 9 * 16 * 32; idx += 512) {
      int s = idx >> 9;
      int rem = idx & 511;
      int o = rem >> 5, kk = rem & 31;
      int di = s / 3, dj = s - di * 3;
      wP[s][o][kk] = f2bf(w_in[((o * C_ + kc * 32 + kk) * 3 + di) * 3 + dj]);
    }
    if (tid < (RW_ + 2) * 32) {
      int t = tid >> 5, kk = tid & 31;
      int ii = i0 - 1 + t;
      hr[t][kk] = (ii >= 0 && ii < L_)
          ? h[(b * L_ + ii) * C_ + kc * 32 + kk] : (unsigned short)0;
    }
    __syncthreads();
#pragma unroll
    for (int di = 0; di < 3; ++di) {
      FragBF hf[RW_];
#pragma unroll
      for (int r = 0; r < RW_; ++r) {
        hf[r].q[0] = *(const v4u*)&hr[r + di][8 * hl];
        hf[r].q[1] = *(const v4u*)&hr[r + di][16 + 8 * hl];
      }
#pragma unroll
      for (int dj = 0; dj < 3; ++dj) {
        int s = di * 3 + dj;
        FragBF wf, bb;
        wf.q[0] = *(const v4u*)&wP[s][lo][8 * hl];
        wf.q[1] = *(const v4u*)&wP[s][lo][16 + 8 * hl];
        const unsigned short* brow = &hTpT[jcol + dj][0];
        bb.q[0] = *(const v4u*)&brow[8 * hl];
        bb.q[1] = *(const v4u*)&brow[16 + 8 * hl];
#pragma unroll
        for (int r = 0; r < RW_; ++r) {
          FragBF am;
          am.v = wf.v * hf[r].v;   // packed bf16 multiply (v_pk_mul_bf16)
          acc[r] = __builtin_amdgcn_wmma_f32_16x16x32_bf16(
              false, am.v, false, bb.v, (short)0, acc[r], false, false);
        }
      }
    }
  }
#pragma unroll
  for (int r = 0; r < RW_; ++r) {
    int i = i0 + r;
#pragma unroll
    for (int v = 0; v < 8; ++v) {
      int o = v + 8 * hl;
      float val = acc[r][v] + b_in[o];
      val = val > 0.f ? val : 0.f;
      int idx = ((b * L_ + i) * L_ + jcol) * BBN_ + o;
      y0[idx] = val;
      x16[idx] = f2bf(val);
    }
  }
}

// ---------------- kernel 3: 16->16 3x3 conv (im2col K=144 pad 160) ---------
// One block per output row, 16 waves = 16 j-tiles; patch staged once.
// mode 0: xupd = bf16(relu(conv+b));  mode 1: y0 = relu(conv+b+y0), xupd=bf16
__global__ void __launch_bounds__(512) conv16(
    const unsigned short* __restrict__ xin,
    const float* __restrict__ w,
    const float* __restrict__ bias,
    float* __restrict__ y0,
    unsigned short* __restrict__ xupd,
    int mode) {
  int b = blockIdx.y, i = blockIdx.x;
  int tid = threadIdx.x;
  int wave = tid >> 5, lane = tid & 31;
  int hl = lane >> 4, lo = lane & 15;
  int jcol = wave * 16 + lo;                  // 0..255
  __shared__ unsigned short xl[3][258][16];   // channel-last patch (rows 32B)
  __shared__ unsigned short wl[16][160];      // [o][c+16*s], zero padded
  for (int idx = tid; idx < 3 * 258 * 16; idx += 512) {
    int di = idx / (258 * 16);
    int rem = idx - di * (258 * 16);
    int jj = rem >> 4, c = rem & 15;
    int ii = i + di - 1;
    int j = jj - 1;
    xl[di][jj][c] = (ii >= 0 && ii < L_ && j >= 0 && j < L_)
        ? xin[((b * L_ + ii) * L_ + j) * BBN_ + c] : (unsigned short)0;
  }
  for (int idx = tid; idx < 16 * 160; idx += 512) {
    int o = idx / 160, k = idx - o * 160;
    unsigned short val = 0;
    if (k < 144) {
      int c = k & 15, s = k >> 4;
      int di = s / 3, dj = s - di * 3;
      val = f2bf(w[((o * BBN_ + c) * 3 + di) * 3 + dj]);
    }
    wl[o][k] = val;
  }
  __syncthreads();
  v8f acc = {};
#pragma unroll
  for (int kc = 0; kc < 5; ++kc) {
    FragBF a, bb;
    a.q[0] = *(const v4u*)&wl[lo][kc * 32 + 8 * hl];
    a.q[1] = *(const v4u*)&wl[lo][kc * 32 + 16 + 8 * hl];
    // group 1: K = kc*32 + 8hl + e  -> s = 2*kc, c = 8hl+e (contiguous)
    {
      int s = 2 * kc;
      int di = s / 3, dj = s - di * 3;
      bb.q[0] = *(const v4u*)&xl[di][jcol + dj][8 * hl];
    }
    // group 2: K = kc*32+16+8hl+e -> s = 2*kc+1 (invalid for kc==4 -> zero)
    if (2 * kc + 1 <= 8) {
      int s = 2 * kc + 1;
      int di = s / 3, dj = s - di * 3;
      bb.q[1] = *(const v4u*)&xl[di][jcol + dj][8 * hl];
    } else {
      v4u z = {};
      bb.q[1] = z;
    }
    acc = __builtin_amdgcn_wmma_f32_16x16x32_bf16(
        false, a.v, false, bb.v, (short)0, acc, false, false);
  }
#pragma unroll
  for (int v = 0; v < 8; ++v) {
    int o = v + 8 * hl;
    int idx = ((b * L_ + i) * L_ + jcol) * BBN_ + o;
    float val = acc[v] + bias[o];
    if (mode == 0) {
      val = val > 0.f ? val : 0.f;
      xupd[idx] = f2bf(val);
    } else {
      val += y0[idx];
      val = val > 0.f ? val : 0.f;
      y0[idx] = val;
      xupd[idx] = f2bf(val);
    }
  }
}

// ---------------- kernel 4: 1x1 head --------------------------------------
__global__ void head_k(const float* __restrict__ y0,
                       const float* __restrict__ w_out,
                       const float* __restrict__ b_out,
                       float* __restrict__ out) {
  int b = blockIdx.y, i = blockIdx.x, j = threadIdx.x;
  const float* p = &y0[((b * L_ + i) * L_ + j) * BBN_];
  float s = b_out[0];
#pragma unroll
  for (int o = 0; o < BBN_; ++o) s += p[o] * w_out[o];
  out[(b * L_ + i) * L_ + j] = s;
}

extern "C" void kernel_launch(void* const* d_in, const int* in_sizes, int n_in,
                              void* d_out, int out_size, void* d_ws, size_t ws_size,
                              hipStream_t stream) {
  (void)in_sizes; (void)n_in; (void)out_size; (void)ws_size;
  const float* hidden = (const float*)d_in[0];
  const float* wmask  = (const float*)d_in[1];
  const float* W_nuc  = (const float*)d_in[2];
  const float* b_nuc  = (const float*)d_in[3];
  const float* w_in   = (const float*)d_in[4];
  const float* b_in   = (const float*)d_in[5];
  const float* w1a    = (const float*)d_in[6];
  const float* b1a    = (const float*)d_in[7];
  const float* w1b    = (const float*)d_in[8];
  const float* b1b    = (const float*)d_in[9];
  const float* w2a    = (const float*)d_in[10];
  const float* b2a    = (const float*)d_in[11];
  const float* w2b    = (const float*)d_in[12];
  const float* b2b    = (const float*)d_in[13];
  const float* w_out  = (const float*)d_in[14];
  const float* b_out  = (const float*)d_in[15];
  const int*   ptl    = (const int*)d_in[16];
  const int*   nucid  = (const int*)d_in[17];

  char* ws = (char*)d_ws;
  size_t off = 0;
  auto alloc = [&](size_t bytes) {
    void* p = ws + off;
    off = (off + bytes + 255) & ~size_t(255);
    return p;
  };
  unsigned short* mapb16 = (unsigned short*)alloc(size_t(B_) * L_ * C_ * 2);
  unsigned short* wnb    = (unsigned short*)alloc(size_t(5) * C_ * C_ * 2);
  unsigned short* h      = (unsigned short*)alloc(size_t(B_) * L_ * C_ * 2);
  unsigned short* hT     = (unsigned short*)alloc(size_t(B_) * C_ * L_ * 2);
  float*          y0     = (float*)alloc(size_t(B_) * L_ * L_ * BBN_ * 4);
  unsigned short* x16    = (unsigned short*)alloc(size_t(B_) * L_ * L_ * BBN_ * 2);
  unsigned short* r16    = (unsigned short*)alloc(size_t(B_) * L_ * L_ * BBN_ * 2);

  prep_map<<<dim3(B_), dim3(256), 0, stream>>>(hidden, ptl, (unsigned int*)mapb16);
  int nw = 5 * C_ * C_;
  swz_wnuc<<<dim3(nw / 256), dim3(256), 0, stream>>>(W_nuc, wnb);
  nuc_gemm<<<dim3(12, 16, B_), dim3(128), 0, stream>>>(
      mapb16, wnb, b_nuc, wmask, nucid, h, hT);
  gram_stem<<<dim3(L_ / RW_, B_), dim3(512), 0, stream>>>(h, hT, w_in, b_in, y0, x16);
  conv16<<<dim3(L_, B_), dim3(512), 0, stream>>>(x16, w1a, b1a, y0, r16, 0);
  conv16<<<dim3(L_, B_), dim3(512), 0, stream>>>(r16, w1b, b1b, y0, x16, 1);
  conv16<<<dim3(L_, B_), dim3(512), 0, stream>>>(x16, w2a, b2a, y0, r16, 0);
  conv16<<<dim3(L_, B_), dim3(512), 0, stream>>>(r16, w2b, b2b, y0, x16, 1);
  head_k<<<dim3(L_, B_), dim3(256), 0, stream>>>(y0, w_out, b_out, (float*)d_out);
}
